// FourierMixingParametrized_6073083756838
// MI455X (gfx1250) — matrix operand order repaired and checked by reading, not compiled, not run on hardware
//
#include <hip/hip_runtime.h>

// Problem dims (fixed by the reference)
#define S_DIM 4096
#define B_DIM 4
#define H_DIM 1024
#define TOK   (S_DIM * B_DIM)   // 16384 = GEMM M
#define NN    H_DIM             // 1024  = GEMM N
#define KK    H_DIM             // 1024  = GEMM K (FFT folded into weights)

typedef __attribute__((ext_vector_type(16))) _Float16 v16h;
typedef __attribute__((ext_vector_type(8)))  _Float16 v8h;
typedef __attribute__((ext_vector_type(8)))  float    v8f;
typedef __attribute__((ext_vector_type(4)))  float    v4f;

union AFrag { v16h v; v8h h[2]; };

// ---------------------------------------------------------------------------
// Kernel 1: fold the 2D FFT into the projection weights.
// W_eff[h, nh'*64+k'] = (1/32) * Re( sum_{nh,k} (Wr + i*Wi)[h,nh,k]
//                                   * exp(+2*pi*i*(nh*nh'/16 + k*k'/64)) )
// Positive-sign radix-2 FFT (16x64) per weight row in LDS.
// ---------------------------------------------------------------------------
__global__ __launch_bounds__(64) void weff_build(const float* __restrict__ W,
                                                 _Float16* __restrict__ weff) {
    __shared__ float re[1024];
    __shared__ float im[1024];
    __shared__ float twr[32];
    __shared__ float twi[32];
    const int h = blockIdx.x;
    const int t = threadIdx.x;  // 0..63

    if (t < 32) {
        float ang = 6.283185307179586f * (float)t * (1.0f / 64.0f);  // +i sign
        twr[t] = __cosf(ang);
        twi[t] = __sinf(ang);
    }
    const float* wrow = W + (size_t)h * (2 * H_DIM);
    for (int i = t; i < 1024; i += 64) {
        int nh = i >> 6, k = i & 63;
        re[i] = wrow[nh * 128 + 2 * k];
        im[i] = wrow[nh * 128 + 2 * k + 1];
    }
    __syncthreads();

    // bit-reverse permutation along k (6 bits)
    for (int i = t; i < 1024; i += 64) {
        int k  = i & 63;
        int rk = (int)(__brev((unsigned)k) >> 26);
        if (k < rk) {
            int j = (i & ~63) | rk;
            float a = re[i]; re[i] = re[j]; re[j] = a;
            float b = im[i]; im[i] = im[j]; im[j] = b;
        }
    }
    __syncthreads();

    // 6 radix-2 stages along k (64-point), all 16 rows in parallel
    for (int s = 1; s <= 6; ++s) {
        int m = 1 << s, half = m >> 1;
        int tstep = 64 >> s;  // w_m^pos == w64^(pos*tstep)
        for (int b = t; b < 512; b += 64) {
            int row = b >> 5;
            int bb  = b & 31;
            int grp = bb / half, pos = bb % half;
            int i0 = row * 64 + grp * m + pos;
            int i1 = i0 + half;
            float wr = twr[pos * tstep], wi = twi[pos * tstep];
            float xr = re[i1], xi = im[i1];
            float vr = xr * wr - xi * wi;
            float vi = xr * wi + xi * wr;
            float ur = re[i0], ui = im[i0];
            re[i0] = ur + vr; im[i0] = ui + vi;
            re[i1] = ur - vr; im[i1] = ui - vi;
        }
        __syncthreads();
    }

    // bit-reverse permutation along nh (4 bits)
    for (int i = t; i < 1024; i += 64) {
        int nh  = i >> 6;
        int rnh = (int)(__brev((unsigned)nh) >> 28);
        if (nh < rnh) {
            int j = rnh * 64 + (i & 63);
            float a = re[i]; re[i] = re[j]; re[j] = a;
            float b = im[i]; im[i] = im[j]; im[j] = b;
        }
    }
    __syncthreads();

    // 4 radix-2 stages along nh (16-point); w16^pos = w64^(4*pos)
    for (int s = 1; s <= 4; ++s) {
        int m = 1 << s, half = m >> 1;
        int tstep = 64 >> s;
        for (int b = t; b < 512; b += 64) {
            int k   = b & 63;
            int bb  = b >> 6;  // 0..7
            int grp = bb / half, pos = bb % half;
            int r0 = grp * m + pos;
            int i0 = r0 * 64 + k;
            int i1 = i0 + half * 64;
            float wr = twr[pos * tstep], wi = twi[pos * tstep];
            float xr = re[i1], xi = im[i1];
            float vr = xr * wr - xi * wi;
            float vi = xr * wi + xi * wr;
            float ur = re[i0], ui = im[i0];
            re[i0] = ur + vr; im[i0] = ui + vi;
            re[i1] = ur - vr; im[i1] = ui - vi;
        }
        __syncthreads();
    }

    // ortho norm over both FFTs: 1/sqrt(16*64) = 1/32; keep real part only
    for (int i = t; i < 1024; i += 64)
        weff[(size_t)h * KK + i] = (_Float16)(re[i] * 0.03125f);
}

// ---------------------------------------------------------------------------
// Kernel 2: streaming f32 -> f16 convert of the activations (one pass).
// ---------------------------------------------------------------------------
__global__ __launch_bounds__(256) void a16_convert(const float* __restrict__ x,
                                                   _Float16* __restrict__ y) {
    size_t i = ((size_t)blockIdx.x * 256 + threadIdx.x) * 8;
    v4f x0 = *(const v4f*)(x + i);
    v4f x1 = *(const v4f*)(x + i + 4);
    v8h o;
#pragma unroll
    for (int e = 0; e < 4; ++e) {
        o[e]     = (_Float16)x0[e];
        o[4 + e] = (_Float16)x1[e];
    }
    *(v8h*)(y + i) = o;
}

// ---------------------------------------------------------------------------
// Kernel 3: out[t,h] = sum_g A16[t,g] * W_eff[h,g] + bias[h]
// M=16384, N=1024, K=1024. Wave tile 64x64 = 4x4 WMMA tiles; pure f16 loads,
// zero VALU in the hot loop -> matrix-issue bound.
// Block: 256 threads = 8 waves, 2(M) x 4(N) -> 128x256 block tile.
// ---------------------------------------------------------------------------
__global__ __launch_bounds__(256) void fft_gemm_f16(const _Float16* __restrict__ A,
                                                    const _Float16* __restrict__ Bw,
                                                    const float* __restrict__ bias,
                                                    float* __restrict__ out) {
    const int lane = threadIdx.x & 31;
    const int wv   = threadIdx.x >> 5;  // 0..7
    const int wm   = wv & 1;
    const int wn   = wv >> 1;
    const int m0   = (blockIdx.x * 2 + wm) * 64;
    const int n0   = (blockIdx.y * 4 + wn) * 64;
    const int lr   = lane & 15;
    const int hi   = lane >> 4;

    const _Float16* arow[4];
    const _Float16* brow[4];
#pragma unroll
    for (int i = 0; i < 4; ++i) arow[i] = A  + (size_t)(m0 + i * 16 + lr) * KK;
#pragma unroll
    for (int j = 0; j < 4; ++j) brow[j] = Bw + (size_t)(n0 + j * 16 + lr) * KK;

    // C layout: VGPR r -> M = r + 8*hi ; lane -> N = lr. Init with bias.
    v8f acc[4][4];
#pragma unroll
    for (int j = 0; j < 4; ++j) {
        float bv = bias[n0 + j * 16 + lr];
#pragma unroll
        for (int i = 0; i < 4; ++i)
#pragma unroll
            for (int r = 0; r < 8; ++r) acc[i][j][r] = bv;
    }

    for (int k0 = 0; k0 < KK; k0 += 32) {
        // B fragment (32x16 f16): lane group hi holds K = hi*16 .. hi*16+15
        AFrag b[4];
#pragma unroll
        for (int j = 0; j < 4; ++j) {
            const _Float16* p = brow[j] + k0 + hi * 16;
            b[j].h[0] = *(const v8h*)(p);
            b[j].h[1] = *(const v8h*)(p + 8);
        }
        // A fragment (16x32 f16): lane group hi holds K = hi*16 .. hi*16+15
        AFrag a[4];
#pragma unroll
        for (int i = 0; i < 4; ++i) {
            const _Float16* p = arow[i] + k0 + hi * 16;
            a[i].h[0] = *(const v8h*)(p);
            a[i].h[1] = *(const v8h*)(p + 8);
            __builtin_prefetch(p + 128, 0, 3);  // stream next K-steps of A
        }
#pragma unroll
        for (int i = 0; i < 4; ++i)
#pragma unroll
            for (int j = 0; j < 4; ++j)
                acc[i][j] = __builtin_amdgcn_wmma_f32_16x16x32_f16(
                    false, a[i].v, false, b[j].v, (short)0, acc[i][j], false, false);
    }

#pragma unroll
    for (int i = 0; i < 4; ++i) {
        int mb = m0 + i * 16 + hi * 8;
#pragma unroll
        for (int j = 0; j < 4; ++j) {
            int n = n0 + j * 16 + lr;
            float* op = out + (size_t)mb * NN + n;
#pragma unroll
            for (int r = 0; r < 8; ++r) op[(size_t)r * NN] = acc[i][j][r];
        }
    }
}

// ---------------------------------------------------------------------------
// Fallback GEMM (A in f32, converted in-kernel): used only if the workspace
// is too small for the f16 activation copy. Wave tile 64x32 = 4x2 tiles.
// ---------------------------------------------------------------------------
__global__ __launch_bounds__(256) void fft_gemm_f32(const float* __restrict__ A,
                                                    const _Float16* __restrict__ Bw,
                                                    const float* __restrict__ bias,
                                                    float* __restrict__ out) {
    const int lane = threadIdx.x & 31;
    const int wv   = threadIdx.x >> 5;
    const int wm   = wv & 1;
    const int wn   = wv >> 1;
    const int m0   = (blockIdx.x * 2 + wm) * 64;
    const int n0   = (blockIdx.y * 4 + wn) * 32;
    const int lr   = lane & 15;
    const int hi   = lane >> 4;

    const float*    arow[4];
    const _Float16* brow[2];
#pragma unroll
    for (int i = 0; i < 4; ++i) arow[i] = A  + (size_t)(m0 + i * 16 + lr) * KK;
#pragma unroll
    for (int j = 0; j < 2; ++j) brow[j] = Bw + (size_t)(n0 + j * 16 + lr) * KK;

    v8f acc[4][2];
#pragma unroll
    for (int j = 0; j < 2; ++j) {
        float bv = bias[n0 + j * 16 + lr];
#pragma unroll
        for (int i = 0; i < 4; ++i)
#pragma unroll
            for (int r = 0; r < 8; ++r) acc[i][j][r] = bv;
    }

    for (int k0 = 0; k0 < KK; k0 += 32) {
        AFrag b[2];
#pragma unroll
        for (int j = 0; j < 2; ++j) {
            const _Float16* p = brow[j] + k0 + hi * 16;
            b[j].h[0] = *(const v8h*)(p);
            b[j].h[1] = *(const v8h*)(p + 8);
        }
        AFrag a[4];
#pragma unroll
        for (int i = 0; i < 4; ++i) {
            const float* p = arow[i] + k0 + hi * 16;
            v4f x0 = *(const v4f*)(p);
            v4f x1 = *(const v4f*)(p + 4);
            v4f x2 = *(const v4f*)(p + 8);
            v4f x3 = *(const v4f*)(p + 12);
#pragma unroll
            for (int e = 0; e < 4; ++e) {
                a[i].v[e]      = (_Float16)x0[e];
                a[i].v[4 + e]  = (_Float16)x1[e];
                a[i].v[8 + e]  = (_Float16)x2[e];
                a[i].v[12 + e] = (_Float16)x3[e];
            }
        }
#pragma unroll
        for (int i = 0; i < 4; ++i)
#pragma unroll
            for (int j = 0; j < 2; ++j)
                acc[i][j] = __builtin_amdgcn_wmma_f32_16x16x32_f16(
                    false, a[i].v, false, b[j].v, (short)0, acc[i][j], false, false);
    }

#pragma unroll
    for (int i = 0; i < 4; ++i) {
        int mb = m0 + i * 16 + hi * 8;
#pragma unroll
        for (int j = 0; j < 2; ++j) {
            int n = n0 + j * 16 + lr;
            float* op = out + (size_t)mb * NN + n;
#pragma unroll
            for (int r = 0; r < 8; ++r) op[(size_t)r * NN] = acc[i][j][r];
        }
    }
}

extern "C" void kernel_launch(void* const* d_in, const int* in_sizes, int n_in,
                              void* d_out, int out_size, void* d_ws, size_t ws_size,
                              hipStream_t stream) {
    (void)in_sizes; (void)n_in; (void)out_size;
    const float* hidden = (const float*)d_in[0];   // [16384,1024] f32
    const float* Wp     = (const float*)d_in[1];   // [1024,2048]  f32
    const float* bp     = (const float*)d_in[2];   // [1024]       f32
    float*    out  = (float*)d_out;                 // [16384,1024] f32
    _Float16* weff = (_Float16*)d_ws;               // 2 MiB

    weff_build<<<dim3(H_DIM), dim3(64), 0, stream>>>(Wp, weff);

    const size_t weff_bytes = (size_t)NN * KK * sizeof(_Float16);   // 2 MiB
    const size_t a16_bytes  = (size_t)TOK * KK * sizeof(_Float16);  // 32 MiB
    if (ws_size >= weff_bytes + a16_bytes) {
        _Float16* a16 = weff + (size_t)NN * KK;
        a16_convert<<<dim3((TOK * KK) / 2048), dim3(256), 0, stream>>>(hidden, a16);
        fft_gemm_f16<<<dim3(TOK / 128, NN / 256), dim3(256), 0, stream>>>(a16, weff, bp, out);
    } else {
        fft_gemm_f32<<<dim3(TOK / 128, NN / 128), dim3(256), 0, stream>>>(hidden, weff, bp, out);
    }
}
